// GptOssMlaAttention_46059229283047
// MI455X (gfx1250) — compile-verified, matrix operand order checked by reading, not run on hardware
//
#include <hip/hip_runtime.h>
#include <cstdint>
#include <cstddef>

#define S_LEN   2048
#define HID_D   2048
#define NH_N    32
#define NOPE_D  128
#define ROPE_D  64
#define QK_D    192
#define V_D     128
#define SW_N    128
#define SCALE_F 0.072168783648703216f   // 192^-0.5
#define NH_QK   (NH_N * QK_D)           // 6144
#define NH_KV   (NH_N * (NOPE_D + V_D)) // 8192
#define NH_VD   (NH_N * V_D)            // 4096
#define CKV_D   (512 + ROPE_D)          // 576

typedef __attribute__((ext_vector_type(16))) __bf16 v16bf;
typedef __attribute__((ext_vector_type(8)))  float  v8f;

union BF16x16 { unsigned int u[8]; v16bf v; };

__device__ __forceinline__ unsigned short f2bf(float f) {
    union { float f; unsigned int u; } x; x.f = f;
    unsigned int r = x.u + 0x7FFFu + ((x.u >> 16) & 1u);   // round-to-nearest-even
    return (unsigned short)(r >> 16);
}
__device__ __forceinline__ float bf2f(unsigned short h) {
    union { unsigned int u; float f; } x; x.u = ((unsigned int)h) << 16;
    return x.f;
}

__device__ __forceinline__ v8f wmma_bf16(const BF16x16& a, const BF16x16& b, v8f c) {
    return __builtin_amdgcn_wmma_f32_16x16x32_bf16(false, a.v, false, b.v,
                                                   (short)0, c, false, false);
}

// --- CDNA5 async copy: 16B per lane, global -> LDS, tracked by ASYNCcnt -----
__device__ __forceinline__ void async_copy16(const void* g, void* lds) {
    unsigned int loff = (unsigned int)(unsigned long long)lds;   // low 32b = LDS byte addr
    unsigned long long gaddr = (unsigned long long)g;
    asm volatile("global_load_async_to_lds_b128 %0, %1, off"
                 :: "v"(loff), "v"(gaddr) : "memory");
}
__device__ __forceinline__ void wait_async0() {
    asm volatile("s_wait_asynccnt 0" ::: "memory");
}

// Load a 16x32 bf16 fragment in ISA A/B layout from a row-major (stride in
// elements) bf16 buffer. lane&15 selects the row (M for A, N for B);
// K pairs are interleaved between lane halves per the ISA table.
__device__ __forceinline__ void load_frag(const unsigned short* base, int stride,
                                          int lane, BF16x16& f) {
    int r  = lane & 15;
    int kh = lane >> 4;
    const unsigned short* p = base + (size_t)r * stride;
#pragma unroll
    for (int v = 0; v < 8; ++v) {
        int kb = (v < 4) ? (kh * 8 + v * 2) : (16 + kh * 8 + (v - 4) * 2);
        f.u[v] = *(const unsigned int*)(p + kb);
    }
}

// V fragment: V stored transposed (NH, V_D, S). K (=key) pairs are contiguous
// along S. kb>=16 (zero-padded half of P) folds back onto the same tile.
__device__ __forceinline__ void load_frag_v(const unsigned short* base, int stride,
                                            int lane, BF16x16& f) {
    int n  = lane & 15;
    int kh = lane >> 4;
    const unsigned short* p = base + (size_t)n * stride;
#pragma unroll
    for (int v = 0; v < 8; ++v) {
        int kb = (v < 4) ? (kh * 8 + v * 2) : (16 + kh * 8 + (v - 4) * 2);
        f.u[v] = *(const unsigned int*)(p + (kb & 15));
    }
}

__device__ __forceinline__ float rmax16(float x) {
#pragma unroll
    for (int off = 1; off < 16; off <<= 1) x = fmaxf(x, __shfl_xor(x, off, 32));
    return x;
}
__device__ __forceinline__ float rsum16(float x) {
#pragma unroll
    for (int off = 1; off < 16; off <<= 1) x += __shfl_xor(x, off, 32);
    return x;
}

__device__ __forceinline__ void store_c(float* p, float v)          { *p = v; }
__device__ __forceinline__ void store_c(unsigned short* p, float v) { *p = f2bf(v); }

// ---------------------------------------------------------------------------
// One-time conversions: fp32 -> bf16, and fp32 (KxN) -> bf16 transposed (NxK)
// ---------------------------------------------------------------------------
__global__ void convert_bf16_kernel(const float* __restrict__ in,
                                    unsigned short* __restrict__ out, int n) {
    int i = blockIdx.x * blockDim.x + threadIdx.x;
    if (i < n) out[i] = f2bf(in[i]);
}

__global__ void transpose_bf16_kernel(const float* __restrict__ B,   // (K,N)
                                      unsigned short* __restrict__ Bt, // (N,K)
                                      int K, int N) {
    __shared__ float tile[32][33];
    int kb = blockIdx.y * 32, nb = blockIdx.x * 32;
    for (int i = threadIdx.y; i < 32; i += 8)
        tile[i][threadIdx.x] = B[(size_t)(kb + i) * N + nb + threadIdx.x];
    __syncthreads();
    for (int i = threadIdx.y; i < 32; i += 8)
        Bt[(size_t)(nb + i) * K + kb + threadIdx.x] = f2bf(tile[threadIdx.x][i]);
}

// ---------------------------------------------------------------------------
// GEMM: C[M,N] = A[M,K] @ Bt[N,K]^T (+ bias). A,Bt bf16; fp32 accumulation.
// 64 x (NT*16) block tile, 128 threads (4 waves, 16 rows each), LDS double
// buffer fed by GLOBAL_LOAD_ASYNC_TO_LDS_B128 (ASYNCcnt), stride-40 padding.
// ---------------------------------------------------------------------------
template <typename CT, int NT>
__global__ __launch_bounds__(128) void gemm_wmma_async(
    const unsigned short* __restrict__ A, int lda,
    const unsigned short* __restrict__ Bt,
    const float* __restrict__ bias,
    CT* __restrict__ C, int N, int K)
{
    constexpr int BN = NT * 16;
    __shared__ unsigned short As[2][64 * 40];
    __shared__ unsigned short Bs[2][BN * 40];

    int tid  = threadIdx.x;
    int lane = tid & 31;
    int wave = tid >> 5;
    int m0 = blockIdx.y * 64;
    int n0 = blockIdx.x * BN;

    v8f acc[NT];
#pragma unroll
    for (int i = 0; i < NT; ++i) acc[i] = (v8f){0, 0, 0, 0, 0, 0, 0, 0};

    auto stage = [&](int buf, int k0) {
        // A tile: 64 rows x 32 bf16 (4 chunks of 16B per row)
#pragma unroll
        for (int i = 0; i < 2; ++i) {
            int e = tid + i * 128;           // 0..255
            int r = e >> 2, c = e & 3;
            async_copy16(A + (size_t)(m0 + r) * lda + k0 + c * 8,
                         &As[buf][r * 40 + c * 8]);
        }
        // B tile: BN rows x 32 bf16
#pragma unroll
        for (int i = 0; i < NT / 2; ++i) {
            int e = tid + i * 128;           // 0..BN*4-1
            int r = e >> 2, c = e & 3;
            async_copy16(Bt + (size_t)(n0 + r) * K + k0 + c * 8,
                         &Bs[buf][r * 40 + c * 8]);
        }
    };

    stage(0, 0);
    wait_async0();
    __syncthreads();

    int cur = 0;
    for (int k0 = 0; k0 < K; k0 += 32) {
        if (k0 + 32 < K) stage(cur ^ 1, k0 + 32);   // prefetch next tile (async)

        BF16x16 af;
        load_frag(&As[cur][wave * 16 * 40], 40, lane, af);
#pragma unroll
        for (int nt = 0; nt < NT; ++nt) {
            BF16x16 bf;
            load_frag(&Bs[cur][nt * 16 * 40], 40, lane, bf);
            acc[nt] = wmma_bf16(af, bf, acc[nt]);
        }

        wait_async0();       // own async copies into buf^1 complete
        __syncthreads();     // all waves done copying and done reading buf
        cur ^= 1;
    }

    int ln = lane & 15, kh = lane >> 4;
#pragma unroll
    for (int nt = 0; nt < NT; ++nt) {
        int n = n0 + nt * 16 + ln;
        float bv = bias ? bias[n] : 0.0f;
#pragma unroll
        for (int v = 0; v < 8; ++v) {
            int m = m0 + wave * 16 + v + 8 * kh;
            store_c(C + (size_t)m * N + n, acc[nt][v] + bv);
        }
    }
}

// ---------------------------------------------------------------------------
// RoPE on q + repack (S,NH,QK) bf16 -> (NH,S,QK) bf16
// ---------------------------------------------------------------------------
__global__ void ropeq_kernel(const unsigned short* __restrict__ qbf,
                             const float* __restrict__ cosb,
                             const float* __restrict__ sinb,
                             unsigned short* __restrict__ Qb)
{
    int idx = blockIdx.x * blockDim.x + threadIdx.x;   // ((h*S)+s)*QK + k
    if (idx >= NH_N * S_LEN * QK_D) return;
    int k = idx % QK_D;
    int rem = idx / QK_D;
    int s = rem % S_LEN;
    int h = rem / S_LEN;
    const unsigned short* qrow = qbf + (size_t)s * NH_QK + h * QK_D;
    float val;
    if (k < NOPE_D) {
        val = bf2f(qrow[k]);
    } else {
        int d = k - NOPE_D;
        float x  = bf2f(qrow[NOPE_D + d]);
        float xr = (d < ROPE_D / 2) ? -bf2f(qrow[NOPE_D + d + ROPE_D / 2])
                                    :  bf2f(qrow[NOPE_D + d - ROPE_D / 2]);
        val = x * cosb[s * ROPE_D + d] + xr * sinb[s * ROPE_D + d];
    }
    Qb[idx] = f2bf(val);
}

// ---------------------------------------------------------------------------
// Build K (NH,S,QK) = [k_nope | roped k_rope] bf16 and V transposed (NH,V_D,S)
// ---------------------------------------------------------------------------
__global__ void packkv_kernel(const unsigned short* __restrict__ kvbf,   // (S, NH*256) bf16
                              const unsigned short* __restrict__ ckvbf,  // (S, 576) bf16
                              const float* __restrict__ cosb,
                              const float* __restrict__ sinb,
                              unsigned short* __restrict__ Kb,
                              unsigned short* __restrict__ Vt)
{
    int idx = blockIdx.x * blockDim.x + threadIdx.x;   // ((h*S)+s)*320 + c
    if (idx >= NH_N * S_LEN * 320) return;
    int c = idx % 320;
    int rem = idx / 320;
    int s = rem % S_LEN;
    int h = rem / S_LEN;
    if (c < NOPE_D) {
        Kb[((size_t)h * S_LEN + s) * QK_D + c] = kvbf[(size_t)s * NH_KV + h * 256 + c];
    } else if (c < QK_D) {
        int d = c - NOPE_D;
        const unsigned short* kr = ckvbf + (size_t)s * CKV_D + 512;
        float x  = bf2f(kr[d]);
        float xr = (d < ROPE_D / 2) ? -bf2f(kr[d + ROPE_D / 2]) : bf2f(kr[d - ROPE_D / 2]);
        float val = x * cosb[s * ROPE_D + d] + xr * sinb[s * ROPE_D + d];
        Kb[((size_t)h * S_LEN + s) * QK_D + c] = f2bf(val);
    } else {
        int d = c - QK_D;                              // 0..127
        Vt[((size_t)h * V_D + d) * S_LEN + s] = kvbf[(size_t)s * NH_KV + h * 256 + NOPE_D + d];
    }
}

// ---------------------------------------------------------------------------
// Sliding-window attention with sink. One wave per (16-query tile, head).
// Flash-style online softmax; sink folded in as m=sink, l=1 initial state.
// ---------------------------------------------------------------------------
__global__ __launch_bounds__(32) void attn_kernel(
    const unsigned short* __restrict__ Qb,   // (NH,S,QK) bf16
    const unsigned short* __restrict__ Kb,   // (NH,S,QK) bf16
    const unsigned short* __restrict__ Vt,   // (NH,V_D,S) bf16
    const float* __restrict__ sinks,
    unsigned short* __restrict__ attO)       // (S, NH*V_D) bf16
{
    __shared__ unsigned short P[16 * 34];    // P tile, K padded to 32 (upper half 0)

    int lane = threadIdx.x;
    int m0 = blockIdx.x * 16;
    int h  = blockIdx.y;
    int ln = lane & 15, kh = lane >> 4;

    for (int i = lane; i < 16 * 34; i += 32) P[i] = 0;
    __syncthreads();

    BF16x16 qf[6];
    const unsigned short* qbase = Qb + ((size_t)h * S_LEN + m0) * QK_D;
#pragma unroll
    for (int c = 0; c < 6; ++c) load_frag(qbase + c * 32, QK_D, lane, qf[c]);

    v8f O[8];
#pragma unroll
    for (int i = 0; i < 8; ++i) O[i] = (v8f){0, 0, 0, 0, 0, 0, 0, 0};

    float sk = sinks[h];
    float mrun[8], lrun[8];
#pragma unroll
    for (int v = 0; v < 8; ++v) { mrun[v] = sk; lrun[v] = 1.0f; }

    int t0 = m0 - (SW_N - 1); if (t0 < 0) t0 = 0; t0 >>= 4;
    int t1 = m0 >> 4;

    for (int t = t0; t <= t1; ++t) {
        // S = Q @ K^T  (16 queries x 16 keys), K = 192 via 6 chained WMMAs
        v8f s = (v8f){0, 0, 0, 0, 0, 0, 0, 0};
        const unsigned short* kbase = Kb + ((size_t)h * S_LEN + t * 16) * QK_D;
#pragma unroll
        for (int c = 0; c < 6; ++c) {
            BF16x16 bf;
            load_frag(kbase + c * 32, QK_D, lane, bf);
            s = wmma_bf16(qf[c], bf, s);
        }

        // mask + scale + online softmax update
        float p[8], alpha[8];
#pragma unroll
        for (int v = 0; v < 8; ++v) {
            int i = m0 + v + 8 * kh;
            int j = t * 16 + ln;
            bool ok = (j <= i) && (i - j < SW_N);
            float sv = ok ? s[v] * SCALE_F : -1e30f;
            float tm = rmax16(sv);
            float mnew = fmaxf(mrun[v], tm);
            alpha[v] = __expf(mrun[v] - mnew);
            float pv = ok ? __expf(sv - mnew) : 0.0f;
            lrun[v] = lrun[v] * alpha[v] + rsum16(pv);
            mrun[v] = mnew;
            p[v] = pv;
        }

        // rescale accumulated output
#pragma unroll
        for (int nt = 0; nt < 8; ++nt)
#pragma unroll
            for (int v = 0; v < 8; ++v) O[nt][v] *= alpha[v];

        // P (C-layout fp32) -> LDS bf16 -> A-fragment
        __syncthreads();
#pragma unroll
        for (int v = 0; v < 8; ++v) P[(v + 8 * kh) * 34 + ln] = f2bf(p[v]);
        __syncthreads();
        BF16x16 pf;
        load_frag(&P[0], 34, lane, pf);

        // O += P @ V  (V_D = 128 -> 8 N-tiles)
#pragma unroll
        for (int nt = 0; nt < 8; ++nt) {
            BF16x16 vf;
            load_frag_v(Vt + ((size_t)h * V_D + nt * 16) * S_LEN + t * 16, S_LEN, lane, vf);
            O[nt] = wmma_bf16(pf, vf, O[nt]);
        }
    }

#pragma unroll
    for (int nt = 0; nt < 8; ++nt)
#pragma unroll
        for (int v = 0; v < 8; ++v) {
            int m = v + 8 * kh;
            attO[(size_t)(m0 + m) * NH_VD + h * V_D + nt * 16 + ln] = f2bf(O[nt][v] / lrun[v]);
        }
}

// ---------------------------------------------------------------------------
extern "C" void kernel_launch(void* const* d_in, const int* in_sizes, int n_in,
                              void* d_out, int out_size, void* d_ws, size_t ws_size,
                              hipStream_t stream)
{
    (void)in_sizes; (void)n_in; (void)out_size; (void)ws_size;

    const float* hidden = (const float*)d_in[0];
    const float* cosb   = (const float*)d_in[1];
    const float* sinb   = (const float*)d_in[2];
    const float* Wq     = (const float*)d_in[3];
    const float* bq     = (const float*)d_in[4];
    const float* Wkva   = (const float*)d_in[5];
    const float* bkva   = (const float*)d_in[6];
    const float* Wkvb   = (const float*)d_in[7];
    const float* Wo     = (const float*)d_in[8];
    const float* bo     = (const float*)d_in[9];
    const float* sinks  = (const float*)d_in[10];
    float* out = (float*)d_out;

    // Workspace layout. Aliases (safe by stream order):
    //   region WqT : WqT (read by GEMM1) -> Kb (written by packkv, later)
    //   region qbf : qbf (read by ropeq) -> attO bf16 (written by attn, later)
    char* p = (char*)d_ws;
    unsigned short* hbf  = (unsigned short*)p; p += (size_t)S_LEN * HID_D * 2;      //  8.4 MB
    unsigned short* WqT  = (unsigned short*)p;
    unsigned short* Kb   = (unsigned short*)p; p += (size_t)NH_QK * HID_D * 2;      // 25.2 MB
    unsigned short* WkvaT= (unsigned short*)p; p += (size_t)CKV_D * HID_D * 2;      //  2.4 MB
    unsigned short* WkvbT= (unsigned short*)p; p += (size_t)NH_KV * 512 * 2;        //  8.4 MB
    unsigned short* WoT  = (unsigned short*)p; p += (size_t)HID_D * NH_VD * 2;      // 16.8 MB
    unsigned short* qbf  = (unsigned short*)p;
    unsigned short* attO = (unsigned short*)p; p += (size_t)S_LEN * NH_QK * 2;      // 25.2 MB
    unsigned short* ckvbf= (unsigned short*)p; p += (size_t)S_LEN * CKV_D * 2;      //  2.4 MB
    unsigned short* kvbf = (unsigned short*)p; p += (size_t)S_LEN * NH_KV * 2;      // 33.6 MB
    unsigned short* Qb   = (unsigned short*)p; p += (size_t)NH_N * S_LEN * QK_D * 2;// 25.2 MB
    unsigned short* Vt   = (unsigned short*)p; p += (size_t)NH_N * V_D * S_LEN * 2; // 16.8 MB

    dim3 t8(32, 8);

    // 0) one-time precision/layout conversion (weights transposed to N x K)
    convert_bf16_kernel<<<(S_LEN * HID_D + 255) / 256, 256, 0, stream>>>(hidden, hbf, S_LEN * HID_D);
    transpose_bf16_kernel<<<dim3(NH_QK / 32, HID_D / 32), t8, 0, stream>>>(Wq,   WqT,   HID_D, NH_QK);
    transpose_bf16_kernel<<<dim3(CKV_D / 32, HID_D / 32), t8, 0, stream>>>(Wkva, WkvaT, HID_D, CKV_D);
    transpose_bf16_kernel<<<dim3(NH_KV / 32, 512 / 32),   t8, 0, stream>>>(Wkvb, WkvbT, 512,   NH_KV);
    transpose_bf16_kernel<<<dim3(HID_D / 32, NH_VD / 32), t8, 0, stream>>>(Wo,   WoT,   NH_VD, HID_D);

    dim3 blk(128);

    // 1) q = hidden @ Wq + bq          (2048 x 2048 x 6144) -> bf16
    gemm_wmma_async<unsigned short, 8><<<dim3(NH_QK / 128, S_LEN / 64), blk, 0, stream>>>(
        hbf, HID_D, WqT, bq, qbf, NH_QK, HID_D);

    // 2) ckv = hidden @ Wkva + bkva    (2048 x 2048 x 576) -> bf16
    gemm_wmma_async<unsigned short, 4><<<dim3(CKV_D / 64, S_LEN / 64), blk, 0, stream>>>(
        hbf, HID_D, WkvaT, bkva, ckvbf, CKV_D, HID_D);

    // 3) kv = kv_lat @ Wkvb            (2048 x 512 x 8192) -> bf16 (A = ckv[:, :512], lda=576)
    gemm_wmma_async<unsigned short, 8><<<dim3(NH_KV / 128, S_LEN / 64), blk, 0, stream>>>(
        ckvbf, CKV_D, WkvbT, nullptr, kvbf, NH_KV, 512);

    // 4) RoPE(q) + repack to (NH,S,QK)   [frees qbf region for attO]
    ropeq_kernel<<<(NH_N * S_LEN * QK_D) / 256, 256, 0, stream>>>(qbf, cosb, sinb, Qb);

    // 5) Build K (nope | roped rope) and V^T   [frees WqT region for Kb]
    packkv_kernel<<<(NH_N * S_LEN * 320 + 255) / 256, 256, 0, stream>>>(
        kvbf, ckvbf, cosb, sinb, Kb, Vt);

    // 6) Sliding-window attention with sink -> bf16 attO
    attn_kernel<<<dim3(S_LEN / 16, NH_N), 32, 0, stream>>>(Qb, Kb, Vt, sinks, attO);

    // 7) out = attO @ Wo + bo          (2048 x 4096 x 2048) -> fp32
    gemm_wmma_async<float, 8><<<dim3(HID_D / 128, S_LEN / 64), blk, 0, stream>>>(
        attO, NH_VD, WoT, bo, out, S_LEN, NH_VD);
}
